// GATConvDGL_66812511256931
// MI455X (gfx1250) — matrix operand order repaired and checked by reading, not compiled, not run on hardware
//
#include <hip/hip_runtime.h>
#include <hip/hip_bf16.h>
#include <stdint.h>

#define NNODES 100000
#define NEDGES 1600000
#define CIN    128
#define COUT   128      // OUT*HEADS
#define NEG_SLOPE 0.2f

typedef __attribute__((ext_vector_type(16))) __bf16 v16bf;
typedef __attribute__((ext_vector_type(8)))  float  v8f;

union Frag16 {             // one WMMA A/B operand (16 bf16 per lane = 32B)
    v16bf v;
    uint4 q[2];
};

__device__ __forceinline__ unsigned short f32_to_bf16_rne(float f) {
    unsigned u = __float_as_uint(f);
    u += 0x7FFFu + ((u >> 16) & 1u);
    return (unsigned short)(u >> 16);
}

// order-preserving float<->uint transform for atomicMax on floats
__device__ __forceinline__ unsigned f2ord(float f) {
    unsigned u = __float_as_uint(f);
    return (u >> 31) ? ~u : (u | 0x80000000u);
}
__device__ __forceinline__ float ord2f(unsigned u) {
    return __uint_as_float((u >> 31) ? (u ^ 0x80000000u) : ~u);
}

__global__ void zero_u32(unsigned* __restrict__ p, int n) {
    int i = blockIdx.x * blockDim.x + threadIdx.x;
    if (i < n) p[i] = 0u;
}

__global__ void cvt_f32_bf16(const float* __restrict__ src,
                             unsigned short* __restrict__ dst, int n) {
    int i = blockIdx.x * blockDim.x + threadIdx.x;
    if (i < n) dst[i] = f32_to_bf16_rne(src[i]);
}

// Z = x @ W^T + b  via v_wmma_f32_16x16x32_bf16.
// grid = 6250 node tiles (16 rows each); block = 256 = 8 waves; wave w owns
// the 16-column tile [w*16, w*16+16). K = 128 in 4 steps of 32.
__global__ __launch_bounds__(256) void gemm_wmma_bf16(
    const unsigned short* __restrict__ xb,   // [N,128] bf16
    const unsigned short* __restrict__ Wb,   // [128,128] bf16 (row n = out channel)
    const float* __restrict__ bias,          // [128]
    float* __restrict__ Z)                   // [N,128]
{
    const int wave = threadIdx.x >> 5;
    const int lane = threadIdx.x & 31;
    const int lm   = lane & 15;
    const int hl   = lane >> 4;
    const int tile = blockIdx.x;
    const int colb = wave * 16;

    const unsigned short* xrow = xb + (size_t)(tile * 16 + lm) * CIN;  // A row (node)
    const unsigned short* wrow = Wb + (size_t)(colb + lm) * CIN;       // B col n = W row

    const float bv = bias[colb + lm];
    v8f acc = { bv, bv, bv, bv, bv, bv, bv, bv };

#pragma unroll
    for (int ks = 0; ks < 4; ++ks) {
        const int k0 = ks * 32;
        Frag16 a, b;
        // A 16x32 bf16 layout: lanes 0-15 hold K 0..7 & 16..23, lanes 16-31 hold 8..15 & 24..31
        a.q[0] = *(const uint4*)(xrow + k0 + hl * 8);
        a.q[1] = *(const uint4*)(xrow + k0 + 16 + hl * 8);
        // B 32x16 bf16 layout: lanes 0-15 hold K 0..7 & 16..23, lanes 16-31 hold 8..15 & 24..31
        b.q[0] = *(const uint4*)(wrow + k0 + hl * 8);
        b.q[1] = *(const uint4*)(wrow + k0 + 16 + hl * 8);
        acc = __builtin_amdgcn_wmma_f32_16x16x32_bf16(
                  false, a.v, false, b.v, (short)0, acc, false, false);
    }

    // D layout: vgpr r, lane -> element (m = r + hl*8, n = lm)
    float* zt = Z + (size_t)(tile * 16 + hl * 8) * COUT + colb + lm;
#pragma unroll
    for (int r = 0; r < 8; ++r)
        zt[(size_t)r * COUT] = acc[r];
}

// e_l[n,h] = sum_c (c%4==h) Z[n,c]*a_l[c] ; one wave per node
__global__ __launch_bounds__(256) void node_scores(
    const float* __restrict__ Z, const float* __restrict__ a_l,
    const float* __restrict__ a_r,
    float* __restrict__ e_l, float* __restrict__ e_r)
{
    const int wave = threadIdx.x >> 5;
    const int lane = threadIdx.x & 31;
    const int node = blockIdx.x * 8 + wave;
    if (node >= NNODES) return;
    const float* zr = Z + (size_t)node * COUT;
    float sl = 0.f, sr = 0.f;
#pragma unroll
    for (int j = 0; j < 4; ++j) {
        const int c = lane + 32 * j;      // c % 4 == lane % 4 for all j
        const float z = zr[c];
        sl += z * a_l[c];
        sr += z * a_r[c];
    }
#pragma unroll
    for (int off = 4; off < 32; off <<= 1) {
        sl += __shfl_xor(sl, off, 32);
        sr += __shfl_xor(sr, off, 32);
    }
    if (lane < 4) {
        e_l[node * 4 + lane] = sl;
        e_r[node * 4 + lane] = sr;
    }
}

__global__ void edge_logits(const int* __restrict__ row, const int* __restrict__ col,
                            const float* __restrict__ e_l, const float* __restrict__ e_r,
                            float* __restrict__ e, unsigned* __restrict__ m_bits)
{
    const int i = blockIdx.x * blockDim.x + threadIdx.x;
    if (i >= NEDGES) return;
    const int r = row[i], c = col[i];
    const float4 el = *(const float4*)(e_l + (size_t)r * 4);
    const float4 er = *(const float4*)(e_r + (size_t)c * 4);
    float4 v;
    v.x = el.x + er.x; v.x = v.x > 0.f ? v.x : NEG_SLOPE * v.x;
    v.y = el.y + er.y; v.y = v.y > 0.f ? v.y : NEG_SLOPE * v.y;
    v.z = el.z + er.z; v.z = v.z > 0.f ? v.z : NEG_SLOPE * v.z;
    v.w = el.w + er.w; v.w = v.w > 0.f ? v.w : NEG_SLOPE * v.w;
    *(float4*)(e + (size_t)i * 4) = v;
    unsigned* mb = m_bits + (size_t)r * 4;
    atomicMax(mb + 0, f2ord(v.x));
    atomicMax(mb + 1, f2ord(v.y));
    atomicMax(mb + 2, f2ord(v.z));
    atomicMax(mb + 3, f2ord(v.w));
}

__global__ void edge_exp(const int* __restrict__ row, const unsigned* __restrict__ m_bits,
                         float* __restrict__ e, float* __restrict__ denom)
{
    const int i = blockIdx.x * blockDim.x + threadIdx.x;
    if (i >= NEDGES) return;
    const int r = row[i];
    const unsigned* mb = m_bits + (size_t)r * 4;
    float4 v = *(float4*)(e + (size_t)i * 4);
    v.x = __expf(v.x - ord2f(mb[0]));
    v.y = __expf(v.y - ord2f(mb[1]));
    v.z = __expf(v.z - ord2f(mb[2]));
    v.w = __expf(v.w - ord2f(mb[3]));
    *(float4*)(e + (size_t)i * 4) = v;
    float* dn = denom + (size_t)r * 4;
    atomicAdd(dn + 0, v.x);
    atomicAdd(dn + 1, v.y);
    atomicAdd(dn + 2, v.z);
    atomicAdd(dn + 3, v.w);
}

// out[row, c] += alpha[e, c%4] * Z[col, c] ; one wave per edge, lane handles
// columns lane, lane+32, lane+64, lane+96 (all same head = lane%4).
__global__ __launch_bounds__(256) void edge_aggregate(
    const int* __restrict__ row, const int* __restrict__ col,
    const float* __restrict__ ex, const float* __restrict__ denom,
    const float* __restrict__ Z, float* __restrict__ out)
{
    const int wave = threadIdx.x >> 5;
    const int lane = threadIdx.x & 31;
    const long i = (long)blockIdx.x * 8 + wave;
    if (i >= NEDGES) return;
    const int r = row[i], c = col[i];
    const int h = lane & 3;
    const float alpha = ex[(size_t)i * 4 + h] / denom[(size_t)r * 4 + h];
    const float* zc = Z + (size_t)c * COUT;
    float* orow = out + (size_t)r * COUT;
#pragma unroll
    for (int j = 0; j < 4; ++j) {
        const int ci = lane + 32 * j;
        atomicAdd(orow + ci, alpha * zc[ci]);
    }
}

extern "C" void kernel_launch(void* const* d_in, const int* in_sizes, int n_in,
                              void* d_out, int out_size, void* d_ws, size_t ws_size,
                              hipStream_t stream) {
    const float* x   = (const float*)d_in[0];
    const int*   row = (const int*)  d_in[1];
    const int*   col = (const int*)  d_in[2];
    const float* W   = (const float*)d_in[3];
    const float* b   = (const float*)d_in[4];
    const float* a_l = (const float*)d_in[5];
    const float* a_r = (const float*)d_in[6];
    float* out = (float*)d_out;

    char* ws = (char*)d_ws;
    size_t off = 0;
    auto alloc = [&](size_t bytes) -> char* {
        char* p = ws + off;
        off = (off + bytes + 255) & ~(size_t)255;
        return p;
    };
    unsigned short* xb    = (unsigned short*)alloc((size_t)NNODES * CIN * 2);
    unsigned short* Wb    = (unsigned short*)alloc((size_t)COUT * CIN * 2);
    float*          Z     = (float*)alloc((size_t)NNODES * COUT * 4);
    float*          e_l   = (float*)alloc((size_t)NNODES * 4 * 4);
    float*          e_r   = (float*)alloc((size_t)NNODES * 4 * 4);
    unsigned*       mbits = (unsigned*)alloc((size_t)NNODES * 4 * 4);   // contiguous with
    float*          denom = (float*)alloc((size_t)NNODES * 4 * 4);      // denom: zero both at once
    float*          ebuf  = (float*)alloc((size_t)NEDGES * 4 * 4);

    // init: out = 0 ; m_bits = 0 (== ord(-inf-ish)) ; denom = 0
    zero_u32<<<(NNODES * COUT + 255) / 256, 256, 0, stream>>>((unsigned*)out, NNODES * COUT);
    zero_u32<<<(NNODES * 8 + 255) / 256, 256, 0, stream>>>(mbits, NNODES * 8);

    // bf16 conversions
    cvt_f32_bf16<<<(NNODES * CIN + 255) / 256, 256, 0, stream>>>(x, xb, NNODES * CIN);
    cvt_f32_bf16<<<(COUT * CIN + 255) / 256, 256, 0, stream>>>(W, Wb, COUT * CIN);

    // Z = x @ W^T + b (WMMA bf16, f32 accumulate)
    gemm_wmma_bf16<<<NNODES / 16, 256, 0, stream>>>(xb, Wb, b, Z);

    // per-node attention scores
    node_scores<<<(NNODES + 7) / 8, 256, 0, stream>>>(Z, a_l, a_r, e_l, e_r);

    // edge softmax (logits -> max -> exp/denominator)
    edge_logits<<<(NEDGES + 255) / 256, 256, 0, stream>>>(row, col, e_l, e_r, ebuf, mbits);
    edge_exp<<<(NEDGES + 255) / 256, 256, 0, stream>>>(row, mbits, ebuf, denom);

    // aggregation
    edge_aggregate<<<(NEDGES + 7) / 8, 256, 0, stream>>>(row, col, ebuf, denom, Z, out);
}
